// DeepPerspectiveNet_69801808495387
// MI455X (gfx1250) — matrix operand order repaired and checked by reading, not compiled, not run on hardware
//
#include <hip/hip_runtime.h>
#include <hip/hip_bf16.h>
#include <math.h>

typedef _Float16 v16h __attribute__((ext_vector_type(16)));
typedef _Float16 v8h  __attribute__((ext_vector_type(8)));
typedef float    v8f  __attribute__((ext_vector_type(8)));

#define BATCH   16384
#define MAXP    32
#define NFEAT   768
#define FTOUT   512
#define L2DIM   32

#define POS_PER_BLOCK 64
#define THREADS       256
#define NBLOCKS       (BATCH / POS_PER_BLOCK)   // 256

// workspace layout (bytes)
#define WPT16_OFF 0                      // 768*512 f16 = 786432 B
#define WL2B_OFF  (NFEAT * FTOUT * 2)    // 32*2*32*16 f16 = 65536 B

// LDS layout (bytes)
#define S_STM   0        // 2048 ints (8 KB) -- aliased as hTile[64][32] f32 after FT
#define S_NSTM  8192     // 2048 ints
#define S_VAL   16384    // 2048 floats
#define S_X     24576    // 4 tiles * 16 * 1024 halves = 128 KB
#define SMEM_BYTES (24576 + 131072)      // 152 KB

union H2  { unsigned int u; _Float16 h[2]; };
union A16 { v16h v; v8h h[2]; };

// ---- prep 1: W_p [512][768] f32 -> WpT [768][512] f16 (feature rows contiguous)
__global__ void prep_wp(const float* __restrict__ Wp, _Float16* __restrict__ WpT) {
    int e = blockIdx.x * 256 + threadIdx.x;
    if (e >= NFEAT * FTOUT) return;
    int f = e >> 9;          // 0..767
    int o = e & 511;         // 0..511
    WpT[e] = (_Float16)Wp[o * NFEAT + f];
}

// ---- prep 2: W_l2 [32][1024] f32 -> B-operand swizzle for v_wmma_f32_16x16x32_f16.
// Layout: for (kk, ntile, lane): 16 contiguous halves = B[k][n] with
// n = ntile*16 + (lane&15), k = kk*32 + (lane>=16 ? 16 : 0) + j, j=0..15.
__global__ void prep_wl2(const float* __restrict__ Wl2, _Float16* __restrict__ B) {
    int e = blockIdx.x * 256 + threadIdx.x;
    if (e >= 32 * 2 * 32 * 16) return;
    int j    = e & 15;
    int lane = (e >> 4) & 31;
    int nt   = (e >> 9) & 1;
    int kk   = e >> 10;
    int n = nt * 16 + (lane & 15);
    int k = kk * 32 + ((lane >> 4) * 16) + j;
    B[e] = (_Float16)Wl2[n * 1024 + k];
}

__device__ __forceinline__ float clamp01(float x) {
    return fminf(fmaxf(x, 0.0f), 1.0f);
}

__global__ __launch_bounds__(THREADS)
void nnue_fused(const int*      __restrict__ stmIdx,
                const int*      __restrict__ nstmIdx,
                const float*    __restrict__ values,
                const _Float16* __restrict__ WpT,
                const float*    __restrict__ bp,
                const _Float16* __restrict__ Wl2B,
                const float*    __restrict__ bl2,
                const float*    __restrict__ Wout,
                const float*    __restrict__ bout,
                float*          __restrict__ out)
{
    __shared__ __align__(16) unsigned char smem[SMEM_BYTES];
    int*   sStm  = (int*)  &smem[S_STM];
    int*   sNstm = (int*)  &smem[S_NSTM];
    float* sVal  = (float*)&smem[S_VAL];
    float* hTile = (float*)&smem[S_STM];   // alias: valid after FT barrier

    const int tid   = threadIdx.x;
    const int lane  = tid & 31;
    const int wave  = tid >> 5;            // 0..7
    const int pBase = blockIdx.x * POS_PER_BLOCK;

    // ---- stage sparse indices + values into LDS (feature = interleaved pair [2g+1])
    for (int e = tid; e < POS_PER_BLOCK * MAXP; e += THREADS) {
        int g = (pBase + (e >> 5)) * MAXP + (e & 31);
        sStm[e]  = stmIdx[2 * g + 1];
        sNstm[e] = nstmIdx[2 * g + 1];
        sVal[e]  = values[g];
    }
    __syncthreads();

    // ---- Phase 1: feature transformer. Wave w owns positions w*8 .. w*8+7.
    // Each lane accumulates 8 consecutive outputs (one uint4 = 8 f16 weights per gather).
    const uint4* Wq = (const uint4*)WpT;   // 64 uint4 per feature row (512 halves)
    for (int pl = 0; pl < 8; ++pl) {
        const int pLocal = wave * 8 + pl;
        const float* vrow = &sVal[pLocal * MAXP];
        for (int persp = 0; persp < 2; ++persp) {
            const int* irow = persp ? &sNstm[pLocal * MAXP] : &sStm[pLocal * MAXP];
            for (int c = 0; c < 2; ++c) {       // 256 outputs per chunk
                const int ob = c * 256 + lane * 8;
                float acc[8];
                #pragma unroll
                for (int t = 0; t < 8; ++t) acc[t] = bp[ob + t];
                for (int j = 0; j < MAXP; ++j) {
                    const int   f  = irow[j];
                    const float vv = vrow[j];
                    uint4 w = Wq[f * 64 + c * 32 + lane];
                    H2 h0, h1, h2, h3;
                    h0.u = w.x; h1.u = w.y; h2.u = w.z; h3.u = w.w;
                    acc[0] = fmaf(vv, (float)h0.h[0], acc[0]);
                    acc[1] = fmaf(vv, (float)h0.h[1], acc[1]);
                    acc[2] = fmaf(vv, (float)h1.h[0], acc[2]);
                    acc[3] = fmaf(vv, (float)h1.h[1], acc[3]);
                    acc[4] = fmaf(vv, (float)h2.h[0], acc[4]);
                    acc[5] = fmaf(vv, (float)h2.h[1], acc[5]);
                    acc[6] = fmaf(vv, (float)h3.h[0], acc[6]);
                    acc[7] = fmaf(vv, (float)h3.h[1], acc[7]);
                }
                v8h o;
                #pragma unroll
                for (int t = 0; t < 8; ++t) {
                    float x = clamp01(acc[t]);
                    o[t] = (_Float16)(x * x);   // clipped-ReLU squared
                }
                // x tile, row-major halves: [pLocal][k], k = persp*512 + c*256 + lane*8
                const int kIdx = persp * 512 + c * 256 + lane * 8;
                *(v8h*)&smem[S_X + (pLocal * 1024 + kIdx) * 2] = o;
            }
        }
    }
    __syncthreads();

    // ---- Phase 2: L2 layer GEMM [16,1024] x [1024,16] per wave via WMMA f16.
    // wave -> (tile = wave>>1, ntile = wave&1); 8 waves = 4 tiles * 2 ntiles.
    {
        const int tile = wave >> 1;
        const int nt   = wave & 1;
        const int m    = lane & 15;
        const int hi   = lane >> 4;        // 0: K{0..15}; 1: K{16..31}
        v8f c = {0.f, 0.f, 0.f, 0.f, 0.f, 0.f, 0.f, 0.f};
        const v8h* Bq = (const v8h*)Wl2B;
        for (int kk = 0; kk < 32; ++kk) {
            A16 A, B;
            const unsigned baseA =
                S_X + (((unsigned)(tile * 16 + m) * 1024u) + (unsigned)(kk * 32 + hi * 16)) * 2u;
            A.h[0] = *(const v8h*)&smem[baseA];        // K offsets 0..7 / 16..23
            A.h[1] = *(const v8h*)&smem[baseA + 16];   // K offsets 8..15 / 24..31
            const int bb = ((kk * 2 + nt) * 32 + lane) * 2;  // in v8h units
            B.h[0] = Bq[bb];
            B.h[1] = Bq[bb + 1];
            c = __builtin_amdgcn_wmma_f32_16x16x32_f16(
                    /*neg_a=*/false, A.v, /*neg_b=*/false, B.v,
                    /*c_mod=*/(short)0, c, /*reuse_a=*/false, /*reuse_b=*/false);
        }
        // epilogue: + b_l2, clip [0,1]; C/D layout: VGPR r -> M = r + hi*8, N = nt*16 + lane%16
        const int n = nt * 16 + (lane & 15);
        const float bl = bl2[n];
        #pragma unroll
        for (int r = 0; r < 8; ++r) {
            const int mr = r + hi * 8;
            float v = clamp01(c[r] + bl);
            hTile[(tile * 16 + mr) * 32 + n] = v;
        }
    }
    __syncthreads();

    // ---- Phase 3: output layer (32 -> 1) + sigmoid
    if (tid < POS_PER_BLOCK) {
        float s = bout[0];
        #pragma unroll
        for (int n = 0; n < L2DIM; ++n)
            s = fmaf(hTile[tid * 32 + n], Wout[n], s);
        out[pBase + tid] = 1.0f / (1.0f + __expf(-s));
    }
}

extern "C" void kernel_launch(void* const* d_in, const int* in_sizes, int n_in,
                              void* d_out, int out_size, void* d_ws, size_t ws_size,
                              hipStream_t stream) {
    const int*   stm    = (const int*)  d_in[0];
    const int*   nstm   = (const int*)  d_in[1];
    const float* values = (const float*)d_in[2];
    // d_in[3] = size scalar (16384), baked in as BATCH
    const float* Wp   = (const float*)d_in[4];
    const float* bp   = (const float*)d_in[5];
    const float* Wl2  = (const float*)d_in[6];
    const float* bl2  = (const float*)d_in[7];
    const float* Wout = (const float*)d_in[8];
    const float* bout = (const float*)d_in[9];

    _Float16* WpT  = (_Float16*)((char*)d_ws + WPT16_OFF);
    _Float16* Wl2B = (_Float16*)((char*)d_ws + WL2B_OFF);

    prep_wp <<<(NFEAT * FTOUT + 255) / 256, 256, 0, stream>>>(Wp, WpT);
    prep_wl2<<<(32 * 2 * 32 * 16 + 255) / 256, 256, 0, stream>>>(Wl2, Wl2B);
    nnue_fused<<<NBLOCKS, THREADS, 0, stream>>>(
        stm, nstm, values, WpT, bp, Wl2B, bl2, Wout, bout, (float*)d_out);
}
